// TransformerBlock_60730837566377
// MI455X (gfx1250) — compile-verified, matrix operand order checked by reading, not run on hardware
//
#include <hip/hip_runtime.h>
#include <math.h>

// ---------------- constants ----------------
#define TT   1024        // tokens
#define DD   1024        // model dim
#define HH   16          // heads
#define HD   64          // head dim
#define EE   8           // experts
#define FF   2560        // expert hidden

typedef __attribute__((ext_vector_type(16))) __bf16 v16bf;
typedef __attribute__((ext_vector_type(8)))  float  v8f;

__device__ __forceinline__ unsigned short f2bfbits(float f) {
    union { float f; unsigned u; } c; c.f = f;
    unsigned u = c.u;
    u += 0x7FFFu + ((u >> 16) & 1u);      // round-to-nearest-even
    return (unsigned short)(u >> 16);
}

// ---------------- LayerNorm ----------------
__global__ void ln_kernel(const float* __restrict__ x, const float* __restrict__ w,
                          const float* __restrict__ b, float* __restrict__ out) {
    int t = blockIdx.x;
    const float* xr = x + (size_t)t * DD;
    __shared__ float red[256];
    int tid = threadIdx.x;
    float s = 0.f, s2 = 0.f;
    for (int d = tid; d < DD; d += 256) { float v = xr[d]; s += v; s2 += v * v; }
    red[tid] = s; __syncthreads();
    for (int o = 128; o > 0; o >>= 1) { if (tid < o) red[tid] += red[tid + o]; __syncthreads(); }
    float mean = red[0] * (1.0f / DD); __syncthreads();
    red[tid] = s2; __syncthreads();
    for (int o = 128; o > 0; o >>= 1) { if (tid < o) red[tid] += red[tid + o]; __syncthreads(); }
    float var = red[0] * (1.0f / DD) - mean * mean;
    float inv = rsqrtf(var + 1e-5f);
    for (int d = tid; d < DD; d += 256)
        out[(size_t)t * DD + d] = (xr[d] - mean) * inv * w[d] + b[d];
}

// ---------------- bf16-WMMA GEMM, 64x64 block tile, LDS-staged ----------------
// C[M,N] = A[M,K] @ B[K,N] + bias (+ residual); gated store for MoE top-1.
// 8 waves / block; each wave owns one 16-row subtile x two 16-col subtiles
// (shared A fragment, 2 WMMAs per K-step). A/B panels are staged in LDS
// pre-swizzled into the ISA 7.12.2 16-bit fragment layout so each lane's
// fragment is one aligned 32-byte LDS read.
__global__ void __launch_bounds__(256)
gemm_wmma(const float* __restrict__ A, const float* __restrict__ B,
          const float* __restrict__ bias, const float* __restrict__ residual,
          float* __restrict__ C, const int* __restrict__ gate_idx, int expert,
          int M, int N, int K) {
    __shared__ __align__(32) unsigned short As[4 * 32 * 16];   // [m_sub][lane][slot]
    __shared__ __align__(32) unsigned short Bs[4 * 32 * 16];   // [n_sub][lane][slot]
    __shared__ int flag;

    int tid = threadIdx.x;
    int m0  = blockIdx.y * 64;
    int n0  = blockIdx.x * 64;

    if (gate_idx) {                                 // block-uniform early exit
        if (tid == 0) flag = 0;
        __syncthreads();
        if (tid < 64 && gate_idx[m0 + tid] == expert) flag = 1;
        __syncthreads();
        if (!flag) return;
    }

    int wave = tid >> 5, lane = tid & 31;
    int half = lane >> 4, r = lane & 15;
    int m_sub  = wave & 3;
    int n_sub0 = (wave >> 2) * 2;

    // cooperative-loader coordinates
    int a_row = tid >> 2;                 // 0..63
    int a_k   = (tid & 3) * 8;            // 0,8,16,24
    int b_row = tid >> 3;                 // 0..31 (k index)
    int b_n   = (tid & 7) * 8;            // 0..56

    // A destination: this thread's 8 k-consecutive elements map to 8
    // consecutive slots of a single fragment lane -> one 16B LDS store.
    int a_lane = (a_row & 15) + ((a_k & 8) ? 16 : 0);
    int a_slot = ((a_k & 16) ? 8 : 0);
    unsigned short* a_dst = &As[(((a_row >> 4) * 32) + a_lane) * 16 + a_slot];

    int b_lane_half = (b_row & 8) ? 16 : 0;
    int b_slot      = ((b_row & 16) ? 8 : 0) + (b_row & 7);

    v8f acc0 = {}, acc1 = {};

    for (int k0 = 0; k0 < K; k0 += 32) {
        if (k0 + 32 < K) {                 // keep HBM stream ahead of staging
            __builtin_prefetch(A + (size_t)(m0 + a_row) * K + (k0 + 32) + a_k, 0, 0);
            __builtin_prefetch(B + (size_t)(k0 + 32 + b_row) * N + n0 + b_n, 0, 0);
        }
        __syncthreads();
        {   // stage A 64x32 panel (coalesced float4 loads, one 16B LDS store)
            const float4* p = reinterpret_cast<const float4*>(
                A + (size_t)(m0 + a_row) * K + k0 + a_k);
            float4 f0 = p[0], f1 = p[1];
            union { unsigned short s[8]; uint4 v; } pk;
            pk.s[0] = f2bfbits(f0.x); pk.s[1] = f2bfbits(f0.y);
            pk.s[2] = f2bfbits(f0.z); pk.s[3] = f2bfbits(f0.w);
            pk.s[4] = f2bfbits(f1.x); pk.s[5] = f2bfbits(f1.y);
            pk.s[6] = f2bfbits(f1.z); pk.s[7] = f2bfbits(f1.w);
            *reinterpret_cast<uint4*>(a_dst) = pk.v;
        }
        {   // stage B 32x64 panel (coalesced float4 loads, scattered b16 stores)
            const float4* p = reinterpret_cast<const float4*>(
                B + (size_t)(k0 + b_row) * N + n0 + b_n);
            float4 f0 = p[0], f1 = p[1];
            float fv[8] = {f0.x, f0.y, f0.z, f0.w, f1.x, f1.y, f1.z, f1.w};
#pragma unroll
            for (int j = 0; j < 8; ++j) {
                int n     = b_n + j;
                int lanei = (n & 15) + b_lane_half;
                Bs[(((n >> 4) * 32) + lanei) * 16 + b_slot] = f2bfbits(fv[j]);
            }
        }
        __syncthreads();
        const v16bf* Af = reinterpret_cast<const v16bf*>(As);
        const v16bf* Bf = reinterpret_cast<const v16bf*>(Bs);
        v16bf a  = Af[m_sub * 32 + lane];
        v16bf b0 = Bf[n_sub0 * 32 + lane];
        v16bf b1 = Bf[(n_sub0 + 1) * 32 + lane];
        acc0 = __builtin_amdgcn_wmma_f32_16x16x32_bf16(
            false, a, false, b0, (short)0, acc0, false, false);
        acc1 = __builtin_amdgcn_wmma_f32_16x16x32_bf16(
            false, a, false, b1, (short)0, acc1, false, false);
    }

#pragma unroll
    for (int i = 0; i < 8; ++i) {
        int orow = m0 + m_sub * 16 + i + half * 8;   // C/D: VGPR i -> M=i+8*half
        if (gate_idx && gate_idx[orow] != expert) continue;
        int c0 = n0 + n_sub0 * 16 + r;
        int c1 = c0 + 16;
        float v0 = acc0[i], v1 = acc1[i];
        if (bias)     { v0 += bias[c0]; v1 += bias[c1]; }
        if (residual) { v0 += residual[(size_t)orow * N + c0];
                        v1 += residual[(size_t)orow * N + c1]; }
        C[(size_t)orow * N + c0] = v0;
        C[(size_t)orow * N + c1] = v1;
    }
}

// ---------------- RoPE (in place on q,k inside qkv buffer) ----------------
__global__ void rope_kernel(float* __restrict__ qkv) {
    int t = blockIdx.x, h = blockIdx.y, i = threadIdx.x;   // i in 0..31
    float inv = powf(10000.0f, -(float)i / 32.0f);          // BASE^-(2i/ROPE)
    float fr  = (float)t * inv;
    float c = cosf(fr), s = sinf(fr);
    float* q = qkv + (size_t)t * (3 * DD) + h * HD;
    float* k = q + DD;
    float q1 = q[i], q2 = q[i + 32];
    q[i] = q1 * c - q2 * s;  q[i + 32] = q1 * s + q2 * c;
    float k1 = k[i], k2 = k[i + 32];
    k[i] = k1 * c - k2 * s;  k[i + 32] = k1 * s + k2 * c;
}

// ---------------- attention: one block per (query, head) ----------------
__global__ void attn_kernel(const float* __restrict__ qkv, const int* __restrict__ ids,
                            float* __restrict__ attn) {
    int qi = blockIdx.x, h = blockIdx.y, tid = threadIdx.x;
    __shared__ float lg[TT];
    __shared__ float red[256];
    __shared__ float part[256];
    const float* qv = qkv + (size_t)qi * (3 * DD) + h * HD;

    float lmax = -3.402823466e38f;
    for (int k = tid; k < TT; k += 256) {
        const float* kv = qkv + (size_t)k * (3 * DD) + DD + h * HD;
        float d = 0.f;
#pragma unroll
        for (int j = 0; j < HD; ++j) d += qv[j] * kv[j];
        d *= 0.125f;                                   // 1/sqrt(64)
        int id = ids[k];
        bool allowed = (k <= qi) || (id >= 2 && id < 8);
        if (!allowed) d = -3.402823466e38f;
        lg[k] = d;
        lmax = fmaxf(lmax, d);
    }
    red[tid] = lmax; __syncthreads();
    for (int o = 128; o > 0; o >>= 1) { if (tid < o) red[tid] = fmaxf(red[tid], red[tid + o]); __syncthreads(); }
    float mx = red[0]; __syncthreads();

    float lsum = 0.f;
    for (int k = tid; k < TT; k += 256) { float e = expf(lg[k] - mx); lg[k] = e; lsum += e; }
    red[tid] = lsum; __syncthreads();
    for (int o = 128; o > 0; o >>= 1) { if (tid < o) red[tid] += red[tid + o]; __syncthreads(); }
    float isum = 1.0f / red[0]; __syncthreads();

    int d = tid & 63, chunk = tid >> 6;
    float a = 0.f;
    for (int k = chunk * 256; k < chunk * 256 + 256; ++k)
        a += lg[k] * qkv[(size_t)k * (3 * DD) + 2 * DD + h * HD + d];
    part[tid] = a; __syncthreads();
    if (tid < 64) {
        float v = part[tid] + part[tid + 64] + part[tid + 128] + part[tid + 192];
        attn[(size_t)qi * DD + h * HD + tid] = v * isum;
    }
}

// ---------------- gate: softmax over 8 experts, top-1 ----------------
__global__ void gate_kernel(const float* __restrict__ h2, const float* __restrict__ gwgt,
                            const float* __restrict__ gb, float* __restrict__ gw,
                            int* __restrict__ gi) {
    int t = blockIdx.x, tid = threadIdx.x;
    __shared__ float red[256][EE];
    float loc[EE] = {};
    for (int d = tid; d < DD; d += 256) {
        float v = h2[(size_t)t * DD + d];
#pragma unroll
        for (int e = 0; e < EE; ++e) loc[e] += v * gwgt[d * EE + e];
    }
#pragma unroll
    for (int e = 0; e < EE; ++e) red[tid][e] = loc[e];
    __syncthreads();
    for (int o = 128; o > 0; o >>= 1) {
        if (tid < o) {
#pragma unroll
            for (int e = 0; e < EE; ++e) red[tid][e] += red[tid + o][e];
        }
        __syncthreads();
    }
    if (tid == 0) {
        float lgv[EE], mx = -3.402823466e38f;
        for (int e = 0; e < EE; ++e) { lgv[e] = red[0][e] + gb[e]; mx = fmaxf(mx, lgv[e]); }
        float s = 0.f;
        for (int e = 0; e < EE; ++e) { lgv[e] = expf(lgv[e] - mx); s += lgv[e]; }
        float best = -1.f; int bi = 0;
        for (int e = 0; e < EE; ++e) { float p = lgv[e] / s; if (p > best) { best = p; bi = e; } }
        gw[t] = best; gi[t] = bi;
    }
}

// ---------------- GLU activation: x1 * gelu(x2), exact gelu ----------------
__global__ void act_kernel(const float* __restrict__ h12, float* __restrict__ act) {
    int idx = blockIdx.x * 256 + threadIdx.x;
    if (idx >= TT * FF) return;
    int t = idx / FF, j = idx - t * FF;
    float x1 = h12[(size_t)t * (2 * FF) + j];
    float x2 = h12[(size_t)t * (2 * FF) + FF + j];
    float g  = 0.5f * x2 * (1.0f + erff(x2 * 0.70710678118654752f));
    act[idx] = x1 * g;
}

// ---------------- MoE aux loss ----------------
__global__ void zero_kernel(float* sums, float* cnts) {
    if (threadIdx.x < EE) { sums[threadIdx.x] = 0.f; cnts[threadIdx.x] = 0.f; }
}
__global__ void stats_kernel(const float* __restrict__ gw, const int* __restrict__ gi,
                             float* sums, float* cnts) {
    int t = blockIdx.x * 256 + threadIdx.x;
    if (t < TT) { atomicAdd(&sums[gi[t]], gw[t]); atomicAdd(&cnts[gi[t]], 1.0f); }
}
__global__ void loss_kernel(const float* sums, const float* cnts, float* out) {
    if (threadIdx.x == 0) {
        float l = 0.f;
        for (int e = 0; e < EE; ++e) {
            float u = sums[e] / (cnts[e] + 1e-8f);
            float d = u - 1.0f / EE;
            l += d * d;
        }
        out[TT * DD] = l;
    }
}

// ---------------- final: out = x1 + outf * gw[token] ----------------
__global__ void final_kernel(const float* __restrict__ x1, const float* __restrict__ outf,
                             const float* __restrict__ gw, float* __restrict__ out) {
    int i = blockIdx.x * 256 + threadIdx.x;
    if (i < TT * DD) out[i] = x1[i] + outf[i] * gw[i >> 10];
}

// ---------------- host launcher ----------------
extern "C" void kernel_launch(void* const* d_in, const int* in_sizes, int n_in,
                              void* d_out, int out_size, void* d_ws, size_t ws_size,
                              hipStream_t stream) {
    const float* x        = (const float*)d_in[0];
    const int*   ids      = (const int*)  d_in[1];
    const float* ln1_w    = (const float*)d_in[2];
    const float* ln1_b    = (const float*)d_in[3];
    const float* qkv_w    = (const float*)d_in[4];
    const float* qkv_b    = (const float*)d_in[5];
    const float* aout_w   = (const float*)d_in[6];
    const float* aout_b   = (const float*)d_in[7];
    const float* ln2_w    = (const float*)d_in[8];
    const float* ln2_b    = (const float*)d_in[9];
    const float* gate_w   = (const float*)d_in[10];
    const float* gate_b   = (const float*)d_in[11];
    const float* fc_w     = (const float*)d_in[12];
    const float* fc_b     = (const float*)d_in[13];
    const float* eout_w   = (const float*)d_in[14];
    const float* eout_b   = (const float*)d_in[15];
    float* out = (float*)d_out;

    // workspace layout (floats)
    float* ws   = (float*)d_ws;
    float* h1   = ws;                           // 1M
    float* qkv  = h1   + (size_t)TT * DD;       // 3M
    float* attn = qkv  + (size_t)TT * 3 * DD;   // 1M
    float* x1   = attn + (size_t)TT * DD;       // 1M
    float* h2   = x1   + (size_t)TT * DD;       // 1M
    float* h12  = h2   + (size_t)TT * DD;       // 5M
    float* act  = h12  + (size_t)TT * 2 * FF;   // 2.5M
    float* outf = act  + (size_t)TT * FF;       // 1M
    float* gw   = outf + (size_t)TT * DD;       // 1K
    int*   gi   = (int*)(gw + TT);              // 1K
    float* sums = (float*)(gi + TT);            // 8
    float* cnts = sums + EE;                    // 8

    // LN1
    ln_kernel<<<TT, 256, 0, stream>>>(x, ln1_w, ln1_b, h1);
    // QKV GEMM: [1024,1024] @ [1024,3072]
    gemm_wmma<<<dim3(3 * DD / 64, TT / 64), 256, 0, stream>>>(
        h1, qkv_w, qkv_b, nullptr, qkv, nullptr, 0, TT, 3 * DD, DD);
    // RoPE on q,k
    rope_kernel<<<dim3(TT, HH), 32, 0, stream>>>(qkv);
    // attention
    attn_kernel<<<dim3(TT, HH), 256, 0, stream>>>(qkv, ids, attn);
    // out-proj GEMM + residual: x1 = x + attn @ aout_w + b
    gemm_wmma<<<dim3(DD / 64, TT / 64), 256, 0, stream>>>(
        attn, aout_w, aout_b, x, x1, nullptr, 0, TT, DD, DD);
    // LN2
    ln_kernel<<<TT, 256, 0, stream>>>(x1, ln2_w, ln2_b, h2);
    // gate
    gate_kernel<<<TT, 256, 0, stream>>>(h2, gate_w, gate_b, gw, gi);
    zero_kernel<<<1, 32, 0, stream>>>(sums, cnts);

    // MoE experts (top-1 routed; second GEMM gated per row, blocks with no
    // routed rows early-exit)
    for (int e = 0; e < EE; ++e) {
        const float* fwe = fc_w   + (size_t)e * DD * 2 * FF;
        const float* fbe = fc_b   + (size_t)e * 2 * FF;
        const float* owe = eout_w + (size_t)e * FF * DD;
        const float* obe = eout_b + (size_t)e * DD;
        gemm_wmma<<<dim3(2 * FF / 64, TT / 64), 256, 0, stream>>>(
            h2, fwe, fbe, nullptr, h12, nullptr, 0, TT, 2 * FF, DD);
        act_kernel<<<(TT * FF + 255) / 256, 256, 0, stream>>>(h12, act);
        gemm_wmma<<<dim3(DD / 64, TT / 64), 256, 0, stream>>>(
            act, owe, obe, nullptr, outf, gi, e, TT, DD, FF);
    }

    stats_kernel<<<(TT + 255) / 256, 256, 0, stream>>>(gw, gi, sums, cnts);
    final_kernel<<<(TT * DD + 255) / 256, 256, 0, stream>>>(x1, outf, gw, out);
    loss_kernel<<<1, 32, 0, stream>>>(sums, cnts, out);
}
